// Attention_77678778515949
// MI455X (gfx1250) — matrix operand order repaired and checked by reading, not compiled, not run on hardware
//
#include <hip/hip_runtime.h>

// MI455X (gfx1250) flash-attention, wave32 + V_WMMA_F32_16X16X32_BF16.
// B=4,H=16,S=2048,D=64 fp32; causal + bias + masks; online softmax (exp2 domain).
// DPP16 butterfly reductions (no ds_bpermute); native bf16 converts.

typedef __attribute__((ext_vector_type(16))) __bf16 v16bf;
typedef __attribute__((ext_vector_type(8)))  float  v8f;

#define S_DIM 2048
#define D_DIM 64
#define H_DIM 16
#define B_DIM 4
#define TQ 64          // q rows per workgroup (16 per wave)
#define TK 64          // keys per LDS tile
#define PITCH 72       // bf16 pitch (144B rows -> 16B aligned, bank-skewed)
#define LOG2E 1.4426950408889634f

__device__ __forceinline__ __bf16 f2bf(float f) { return (__bf16)f; }  // v_cvt bf16

// DPP16 cross-lane move (VALU, no LDS): ctrl is a compile-time DPP control.
template <int CTRL>
__device__ __forceinline__ float dpp_f(float x) {
    return __int_as_float(__builtin_amdgcn_update_dpp(
        0, __float_as_int(x), CTRL, 0xf, 0xf, true));
}
// Butterfly reduction over each 16-lane row (all lanes get the row result).
__device__ __forceinline__ float rmax16(float x) {
    x = fmaxf(x, dpp_f<0xB1>(x));    // quad_perm [1,0,3,2]  (xor 1)
    x = fmaxf(x, dpp_f<0x4E>(x));    // quad_perm [2,3,0,1]  (xor 2)
    x = fmaxf(x, dpp_f<0x141>(x));   // row_half_mirror      (xor 7)
    x = fmaxf(x, dpp_f<0x140>(x));   // row_mirror           (xor 15)
    return x;
}
__device__ __forceinline__ float rsum16(float x) {
    x += dpp_f<0xB1>(x);
    x += dpp_f<0x4E>(x);
    x += dpp_f<0x141>(x);
    x += dpp_f<0x140>(x);
    return x;
}

union AF { v16bf v; __bf16 h[16]; uint4 q[2]; };
union PK4 { __bf16 h[4]; unsigned long long w; };

__global__ __launch_bounds__(128)
void Attention_77678778515949_kernel(const float* __restrict__ Q,
                                     const float* __restrict__ K,
                                     const float* __restrict__ V,
                                     const unsigned char* __restrict__ qmask,
                                     const unsigned char* __restrict__ vmask,
                                     const float* __restrict__ bias,
                                     float* __restrict__ Out)
{
    __shared__ __bf16 Kl[TK * PITCH];        // K tile [key][d]
    __shared__ __bf16 Vt[D_DIM * PITCH];     // V tile transposed [d][key]
    __shared__ __bf16 Pl[4 * 16 * PITCH];    // per-wave P tiles [wave][m][kk]

    const int bh   = blockIdx.y;                 // 0..B*H-1
    const int b    = bh / H_DIM;
    const int q0   = blockIdx.x * TQ;
    const int tid  = threadIdx.x;
    const int wave = tid >> 5;
    const int lane = tid & 31;
    const int half = lane >> 4;                  // 0: lanes 0-15, 1: lanes 16-31
    const int l16  = lane & 15;

    const size_t baseBH = (size_t)bh * S_DIM * D_DIM;
    const float* Qp = Q + baseBH;
    const float* Kp = K + baseBH;
    const float* Vp = V + baseBH;

    const int qbase = q0 + wave * 16;

    // ---- Q A-fragments (16x32 bf16 x2, d=0..63), pre-scaled by log2e/sqrt(D) ----
    AF qa[2];
    {
        const float* qrow = Qp + (size_t)(qbase + l16) * D_DIM;   // A: M = lane&15
        const float qs = 0.125f * LOG2E;
        #pragma unroll
        for (int j = 0; j < 2; ++j) {
            const int d0 = 32 * j + 16 * half;        // chunk 1: K = 16*half + e
            const int d1 = 32 * j + 16 * half + 8;    // chunk 2: K = 16*half+8+e
            #pragma unroll
            for (int e = 0; e < 8; ++e) {
                qa[j].h[e]     = f2bf(qrow[d0 + e] * qs);
                qa[j].h[8 + e] = f2bf(qrow[d1 + e] * qs);
            }
        }
    }

    // ---- accumulators / softmax state (rows r+8*half, replicated across 16 lanes) ----
    v8f o[4];
    #pragma unroll
    for (int dd = 0; dd < 4; ++dd)
        #pragma unroll
        for (int r = 0; r < 8; ++r) o[dd][r] = 0.0f;
    float mrow[8], lrow[8];                      // m in log2 domain
    #pragma unroll
    for (int r = 0; r < 8; ++r) { mrow[r] = -1e30f; lrow[r] = 0.0f; }

    // per-lane bias base: element (r,t) at bp[r*S + kcol]
    const float* bp = bias + ((size_t)b * S_DIM + qbase + 8 * half) * S_DIM;

    const int nkt = q0 / TK + 1;                 // causal: only tiles with k0 <= q_hi
    for (int kt = 0; kt < nkt; ++kt) {
        const int k0 = kt * TK;

        // ---- cooperative K/V tile load: fp32 -> bf16, V transposed ----
        #pragma unroll
        for (int i = 0; i < 8; ++i) {
            const int idx = tid + 128 * i;       // 0..1023 float4 slots
            const int key = idx >> 4;
            const int d   = (idx & 15) << 2;
            const float4 kv = *(const float4*)(Kp + (size_t)(k0 + key) * D_DIM + d);
            const float4 vv = *(const float4*)(Vp + (size_t)(k0 + key) * D_DIM + d);
            PK4 pk;
            pk.h[0] = f2bf(kv.x); pk.h[1] = f2bf(kv.y);
            pk.h[2] = f2bf(kv.z); pk.h[3] = f2bf(kv.w);
            *(unsigned long long*)&Kl[key * PITCH + d] = pk.w;
            Vt[(d + 0) * PITCH + key] = f2bf(vv.x);
            Vt[(d + 1) * PITCH + key] = f2bf(vv.y);
            Vt[(d + 2) * PITCH + key] = f2bf(vv.z);
            Vt[(d + 3) * PITCH + key] = f2bf(vv.w);
        }
        if (kt + 1 < nkt) {                      // prefetch next tile (global_prefetch)
            __builtin_prefetch(Kp + (size_t)(k0 + TK) * D_DIM + tid * 32, 0, 3);
            __builtin_prefetch(Vp + (size_t)(k0 + TK) * D_DIM + tid * 32, 0, 3);
        }
        __syncthreads();

        __bf16* Pw = &Pl[wave * 16 * PITCH];

        // ---- scores: S = (Q*log2e/8) * K^T, all 4 subtiles kept in registers ----
        v8f sc[4];
        #pragma unroll
        for (int t = 0; t < 4; ++t) {
            #pragma unroll
            for (int r = 0; r < 8; ++r) sc[t][r] = 0.0f;
            #pragma unroll
            for (int j = 0; j < 2; ++j) {        // d = 0..31, 32..63
                AF bf;                           // B: N=l16 (key), K=16*half+e (d)
                const uint4* src = (const uint4*)&Kl[(16 * t + l16) * PITCH + 32 * j + 16 * half];
                bf.q[0] = src[0]; bf.q[1] = src[1];
                sc[t] = __builtin_amdgcn_wmma_f32_16x16x32_bf16(
                            false, qa[j].v, false, bf.v, (short)0, sc[t], false, false);
            }
        }

        // ---- bias (log2 domain) + masks in C layout (row r+8*half, col l16) ----
        bool vok[4];
        #pragma unroll
        for (int t = 0; t < 4; ++t)
            vok[t] = vmask[(size_t)bh * S_DIM + (k0 + 16 * t + l16)] != 0;

        if (kt + 1 < nkt) {                      // fully causal-visible tile
            #pragma unroll
            for (int t = 0; t < 4; ++t) {
                const int kcol = k0 + 16 * t + l16;
                #pragma unroll
                for (int r = 0; r < 8; ++r) {
                    const float s = sc[t][r] + bp[(size_t)r * S_DIM + kcol] * LOG2E;
                    sc[t][r] = vok[t] ? s : -1e30f;
                }
            }
        } else {                                 // diagonal tile: per-element causal
            #pragma unroll
            for (int t = 0; t < 4; ++t) {
                const int kcol = k0 + 16 * t + l16;
                #pragma unroll
                for (int r = 0; r < 8; ++r) {
                    const int qrow = qbase + r + 8 * half;
                    const float s = sc[t][r] + bp[(size_t)r * S_DIM + kcol] * LOG2E;
                    sc[t][r] = (vok[t] && (qrow >= kcol)) ? s : -1e30f;
                }
            }
        }

        // ---- single online-softmax update per 64-key tile (exp2 domain) ----
        #pragma unroll
        for (int r = 0; r < 8; ++r) {
            float mx = fmaxf(fmaxf(sc[0][r], sc[1][r]), fmaxf(sc[2][r], sc[3][r]));
            mx = rmax16(mx);
            const float nm   = fmaxf(mrow[r], mx);
            const float corr = exp2f(mrow[r] - nm);
            mrow[r] = nm;
            float psum = 0.0f;
            #pragma unroll
            for (int t = 0; t < 4; ++t) {
                const float p = exp2f(sc[t][r] - nm);
                psum += p;
                Pw[(r + 8 * half) * PITCH + 16 * t + l16] = f2bf(p);
            }
            lrow[r] = lrow[r] * corr + rsum16(psum);
            #pragma unroll
            for (int dd = 0; dd < 4; ++dd) o[dd][r] *= corr;
        }

        // ---- O += P * V  (A from Pw with lane-transpose via LDS, B from Vt) ----
        #pragma unroll
        for (int j = 0; j < 2; ++j) {            // kk = 0..31, 32..63
            AF pa;                               // A: M=l16, K=16*half+e
            pa.q[0] = *(const uint4*)&Pw[l16 * PITCH + 32 * j + 16 * half];
            pa.q[1] = *(const uint4*)&Pw[l16 * PITCH + 32 * j + 16 * half + 8];
            #pragma unroll
            for (int dd = 0; dd < 4; ++dd) {
                AF vb;                           // B: N=16*dd+l16 (d), K=16*half+e (key)
                const uint4* vs = (const uint4*)&Vt[(16 * dd + l16) * PITCH + 32 * j + 16 * half];
                vb.q[0] = vs[0]; vb.q[1] = vs[1];
                o[dd] = __builtin_amdgcn_wmma_f32_16x16x32_bf16(
                            false, pa.v, false, vb.v, (short)0, o[dd], false, false);
            }
        }
        __syncthreads();                         // before next tile overwrites Kl/Vt
    }

    // ---- normalize, apply queries_mask, store fp32 ----
    #pragma unroll
    for (int r = 0; r < 8; ++r) {
        const int qrow = qbase + r + 8 * half;
        const float inv = 1.0f / lrow[r];
        const unsigned char qb = qmask[(size_t)bh * S_DIM + qrow];
        float* orow = Out + baseBH + (size_t)qrow * D_DIM;
        #pragma unroll
        for (int dd = 0; dd < 4; ++dd)
            orow[16 * dd + l16] = qb ? o[dd][r] * inv : 0.0f;
    }
}

extern "C" void kernel_launch(void* const* d_in, const int* in_sizes, int n_in,
                              void* d_out, int out_size, void* d_ws, size_t ws_size,
                              hipStream_t stream) {
    (void)in_sizes; (void)n_in; (void)out_size; (void)d_ws; (void)ws_size;
    const float*         Q  = (const float*)d_in[0];
    const float*         K  = (const float*)d_in[1];
    const float*         V  = (const float*)d_in[2];
    const unsigned char* qm = (const unsigned char*)d_in[3];  // bool -> 1 byte
    const unsigned char* vm = (const unsigned char*)d_in[4];
    const float*         Bs = (const float*)d_in[5];
    float*               O  = (float*)d_out;

    dim3 grid(S_DIM / TQ, B_DIM * H_DIM);   // 32 x 64 workgroups
    dim3 block(128);                        // 4 waves (wave32)
    Attention_77678778515949_kernel<<<grid, block, 0, stream>>>(Q, K, V, qm, vm, Bs, O);
}
